// PiLinear_23373212025220
// MI455X (gfx1250) — matrix operand order repaired and checked by reading, not compiled, not run on hardware
//
#include <hip/hip_runtime.h>

typedef _Float16 h8   __attribute__((ext_vector_type(8)));
typedef _Float16 v16h __attribute__((ext_vector_type(16)));
typedef float    v8f  __attribute__((ext_vector_type(8)));

// pi quantization constants: u = (pi/2)/256
__device__ __constant__ float kPI_UNIT     = 1.5707963267948966f / 256.0f;
__device__ __constant__ float kINV_PI_UNIT = 256.0f / 1.5707963267948966f;

// ---------------------------------------------------------------------------
// Pass 1: fp32 -> integer-grid fp16.  out[i] = (half)rintf(in[i]/u)
// (exact in fp16: |round(x/u)| < 2048 for both x and weight)
// ---------------------------------------------------------------------------
__global__ __launch_bounds__(256) void pi_quant_f32_to_f16(
    const float* __restrict__ in, _Float16* __restrict__ out, long long n8) {
  long long i = (long long)blockIdx.x * blockDim.x + threadIdx.x;
  if (i >= n8) return;
  float4 f0 = ((const float4*)in)[i * 2 + 0];
  float4 f1 = ((const float4*)in)[i * 2 + 1];
  const float s = kINV_PI_UNIT;
  h8 o;
  o[0] = (_Float16)rintf(f0.x * s);
  o[1] = (_Float16)rintf(f0.y * s);
  o[2] = (_Float16)rintf(f0.z * s);
  o[3] = (_Float16)rintf(f0.w * s);
  o[4] = (_Float16)rintf(f1.x * s);
  o[5] = (_Float16)rintf(f1.y * s);
  o[6] = (_Float16)rintf(f1.z * s);
  o[7] = (_Float16)rintf(f1.w * s);
  ((h8*)out)[i] = o;
}

// ---------------------------------------------------------------------------
// Pass 2: WMMA GEMM with double-buffered GLOBAL_LOAD_ASYNC_TO_LDS.
//   C[m,n] = epilogue( sum_k A[m,k] * B[n,k] ),  A:[M,K] f16, B:[N,K] f16.
// Tile: 128(M) x 128(N) x 64(K), 256 threads = 8 wave32.
// Wave grid 2x4: each wave owns 64(M) x 32(N) = 4x2 accumulator frags.
// ---------------------------------------------------------------------------
#define BM 128
#define BN 128
#define BK 64
#define LDT 72            // padded LDS row stride in halves (144 B, 16B-aligned)
#define ABUF (BM * LDT)   // halves per buffer (A and B buffers are equal size)

// Issue one 128 x 64-half tile per operand: 8 async b128 loads per thread.
// gA/gB are pre-offset to (tileRow0 * K + kt).
__device__ __forceinline__ void async_tile(const _Float16* gA, const _Float16* gB,
                                           _Float16* lA, _Float16* lB,
                                           int tid, int K) {
#pragma unroll
  for (int q = 0; q < 4; ++q) {
    int c  = tid * 4 + q;   // 0..1023 chunks of 8 halves (16 B)
    int r  = c >> 3;        // 0..127
    int kc = c & 7;         // 0..7
    unsigned la = (unsigned)(unsigned long long)&lA[r * LDT + kc * 8];
    unsigned lb = (unsigned)(unsigned long long)&lB[r * LDT + kc * 8];
    unsigned long long ga = (unsigned long long)&gA[(size_t)r * K + kc * 8];
    unsigned long long gb = (unsigned long long)&gB[(size_t)r * K + kc * 8];
    asm volatile("global_load_async_to_lds_b128 %0, %1, off"
                 :: "v"(la), "v"(ga) : "memory");
    asm volatile("global_load_async_to_lds_b128 %0, %1, off"
                 :: "v"(lb), "v"(gb) : "memory");
  }
}

__global__ __launch_bounds__(256) void pi_wmma_gemm(
    const _Float16* __restrict__ A, const _Float16* __restrict__ B,
    const float* __restrict__ bias, float* __restrict__ C,
    int M, int N, int K) {
  extern __shared__ _Float16 smem[];
  // Layout: [A buf0][A buf1][B buf0][B buf1], each ABUF halves.
  // NOTE: no arrays of LDS-derived pointers (lld can't relocate such
  // aggregate initializers) — select buffers by pointer arithmetic.

  const int tid  = threadIdx.x;
  const int lane = tid & 31;
  const int wv   = tid >> 5;   // 0..7
  const int wr   = wv >> 2;    // 0..1 -> wave M base = wr*64
  const int wc   = wv & 3;     // 0..3 -> wave N base = wc*32
  const int hl   = lane & 15;
  const int hi   = lane >> 4;  // half-wave select

  const int rowBase = blockIdx.y * BM;
  const int colBase = blockIdx.x * BN;
  const _Float16* gA = A + (size_t)rowBase * K;
  const _Float16* gB = B + (size_t)colBase * K;

  v8f acc[4][2] = {};

  // prologue: fill buffer 0
  async_tile(gA, gB, smem, smem + 2 * ABUF, tid, K);

  int cur = 0;
  for (int kt = 0; kt < K; kt += BK) {
    const bool has_next = (kt + BK) < K;
    if (has_next) {
      // ping-pong: fill the other buffer while this one is consumed
      int nxt = cur ^ 1;
      async_tile(gA + kt + BK, gB + kt + BK,
                 smem + nxt * ABUF, smem + (2 + nxt) * ABUF, tid, K);
      // async loads complete IN ORDER: <=8 outstanding => current tile landed
      asm volatile("s_wait_asynccnt 0x8" ::: "memory");
    } else {
      asm volatile("s_wait_asynccnt 0x0" ::: "memory");
    }
    __syncthreads();   // current tile visible to all waves

    const _Float16* lA = smem + cur * ABUF;
    const _Float16* lB = smem + (2 + cur) * ABUF;

#pragma unroll
    for (int ks = 0; ks < BK; ks += 32) {
      // load ALL fragments first, then issue the 8 WMMAs
      v16h bf[2];
#pragma unroll
      for (int j = 0; j < 2; ++j) {
        // B frag 32x16: lane -> col n = hl, K chunks at hi*8 and hi*8+16
        int n    = wc * 32 + j * 16 + hl;
        int koff = ks + hi * 8;
        h8 lo = *(const h8*)&lB[n * LDT + koff];
        h8 ho = *(const h8*)&lB[n * LDT + koff + 16];
        bf[j] = __builtin_shufflevector(lo, ho, 0, 1, 2, 3, 4, 5, 6, 7, 8, 9,
                                        10, 11, 12, 13, 14, 15);
      }
      v16h af[4];
#pragma unroll
      for (int i = 0; i < 4; ++i) {
        // A frag 16x32: lane -> row m = hl, K chunks at hi*8 and hi*8+16
        int m    = wr * 64 + i * 16 + hl;
        int koff = ks + hi * 8;
        h8 lo = *(const h8*)&lA[m * LDT + koff];
        h8 ho = *(const h8*)&lA[m * LDT + koff + 16];
        af[i] = __builtin_shufflevector(lo, ho, 0, 1, 2, 3, 4, 5, 6, 7, 8, 9,
                                        10, 11, 12, 13, 14, 15);
      }
#pragma unroll
      for (int i = 0; i < 4; ++i)
#pragma unroll
        for (int j = 0; j < 2; ++j)
          acc[i][j] = __builtin_amdgcn_wmma_f32_16x16x32_f16(
              false, af[i], false, bf[j], (short)0, acc[i][j], false, false);
    }

    __syncthreads();   // all waves done reading cur before it is refilled
    cur ^= 1;
  }

  // ---- epilogue: c = round(u*S)*u ; out = round((c + bias)/u)*u
  const float u = kPI_UNIT, s = kINV_PI_UNIT;
#pragma unroll
  for (int i = 0; i < 4; ++i) {
    int mBase = rowBase + wr * 64 + i * 16;
#pragma unroll
    for (int j = 0; j < 2; ++j) {
      int col = colBase + wc * 32 + j * 16 + hl;
      float bv = bias[col];
#pragma unroll
      for (int v = 0; v < 8; ++v) {
        int m = mBase + v + hi * 8;   // C layout: VGPR v -> row v / v+8
        float S  = acc[i][j][v];
        float cq = rintf(S * u) * u;  // pi_quantize(u^2 * S)
        C[(size_t)m * N + col] = rintf((cq + bv) * s) * u;
      }
    }
  }
}

extern "C" void kernel_launch(void* const* d_in, const int* in_sizes, int n_in,
                              void* d_out, int out_size, void* d_ws, size_t ws_size,
                              hipStream_t stream) {
  const float* x  = (const float*)d_in[0];   // [M, IN] fp32
  const float* w  = (const float*)d_in[1];   // [OUT, IN] fp32
  const float* b  = (const float*)d_in[2];   // [OUT] fp32
  float* out = (float*)d_out;                // [M, OUT] fp32

  const int       OUT = in_sizes[2];
  const long long wN  = in_sizes[1];
  const int       IN  = (int)(wN / OUT);
  const long long xN  = in_sizes[0];
  const int       M   = (int)(xN / IN);

  _Float16* xq = (_Float16*)d_ws;            // [M, IN] fp16
  _Float16* wq = xq + (size_t)M * IN;        // [OUT, IN] fp16

  {
    long long n8 = xN / 8;
    pi_quant_f32_to_f16<<<dim3((unsigned)((n8 + 255) / 256)), dim3(256), 0, stream>>>(x, xq, n8);
  }
  {
    long long n8 = wN / 8;
    pi_quant_f32_to_f16<<<dim3((unsigned)((n8 + 255) / 256)), dim3(256), 0, stream>>>(w, wq, n8);
  }

  const size_t lds_bytes = (size_t)4 * ABUF * sizeof(_Float16);  // 73728 B
  dim3 grid(OUT / BN, M / BM);   // N and M are multiples of 128 here
  pi_wmma_gemm<<<grid, dim3(256), lds_bytes, stream>>>(xq, wq, b, out, M, OUT, IN);
}
